// WindowAttention_29214367548031
// MI455X (gfx1250) — matrix operand order repaired and checked by reading, not compiled, not run on hardware
//
#include <hip/hip_runtime.h>
#include <hip/hip_bf16.h>

// ---------------------------------------------------------------------------
// CDNA5 (gfx1250) fused Swin window attention, bf16 WMMA path.
//   B=4096 windows, N=49 tokens (pad 64), DIM=384, HEADS=12, HEAD_DIM=32
// All matmuls use v_wmma_f32_16x16x32_bf16 (wave32, 16x16 tiles, K=32).
// Round 2: LDS layouts transposed + padded (8*odd element strides) so every
// WMMA operand build is two aligned 16B chunks -> ds_load_b128, no u16 storm.
// ---------------------------------------------------------------------------

typedef __attribute__((ext_vector_type(16))) __bf16 v16bf;
typedef __attribute__((ext_vector_type(8)))  __bf16 v8bf;
typedef __attribute__((ext_vector_type(8)))  float  v8f;

typedef unsigned short ushort_t;
typedef unsigned int   uint_t;

#define WIN_N   49
#define NPAD    64
#define DIM     384
#define HEADS   12
#define HDIM    32
#define QKV_CH  1152   // 3*HEADS*HDIM
#define NWIN    4096
#define NMASK   1024

// padded LDS row strides (bf16 elements): multiples of 8 (16B align), /8 odd
#define WSP     392    // weight tile row stride  (384 + 8)
#define XKP     72     // B-panel K stride        (64 + 8)
#define DPP     40     // Q/K token-row stride    (32 + 8)
#define VSP     72     // V row stride            (64 + 8)
#define PSP     72     // P row stride            (64 + 8)
#define SSP     65     // S row stride (f32, odd dwords)

static __device__ __forceinline__ ushort_t f2bf(float f) {
  uint_t u = __float_as_uint(f);
  uint_t r = u + 0x7FFFu + ((u >> 16) & 1u);   // round-to-nearest-even
  return (ushort_t)(r >> 16);
}

static __device__ __forceinline__ v8bf ld8(const ushort_t* p) {
  return *reinterpret_cast<const v8bf*>(p);    // 16B-aligned by construction
}
static __device__ __forceinline__ v16bf cat16(v8bf lo, v8bf hi) {
  return __builtin_shufflevector(lo, hi, 0, 1, 2, 3, 4, 5, 6, 7,
                                         8, 9, 10, 11, 12, 13, 14, 15);
}

static __device__ __forceinline__ v8f wmma_bf16(v16bf a, v16bf b, v8f c) {
  // (neg_a, A, neg_b, B, c_mod, C, reuse_a, reuse_b)
  return __builtin_amdgcn_wmma_f32_16x16x32_bf16(false, a, false, b,
                                                 (short)0, c, false, false);
}

// ---------------------------------------------------------------------------
// Kernel 1: QKV projection.  D = qkv_w (1152x384) x x[b] (384x49) + qkv_b
// grid = (18 row-tiles of 64, 512 window-groups of 8), block = 128 (4 waves)
// Ws: row-major [64][WSP]  (A operand: contiguous K per lane)
// Xs: token-major [64][XKP] (B operand: contiguous K per lane, transposed in)
// Output: channel-major bf16  qkv[b][o][n]
// ---------------------------------------------------------------------------
__global__ __launch_bounds__(128) void qkv_gemm_k(
    const float* __restrict__ x, const float* __restrict__ qkv_w,
    const float* __restrict__ qkv_b, ushort_t* __restrict__ qkv_o) {
  __shared__ ushort_t Ws[64 * WSP];   // ~49 KB
  __shared__ ushort_t Xs[64 * XKP];   //  9 KB, token-major

  const int tid  = threadIdx.x;
  const int lane = tid & 31;
  const int wave = tid >> 5;
  const int half = lane >> 4;
  const int l15  = lane & 15;
  const int M0   = blockIdx.x * 64;

  for (int i = tid; i < 64 * DIM; i += 128) {
    int r = i / DIM, c = i % DIM;
    Ws[r * WSP + c] = f2bf(qkv_w[(size_t)(M0 + r) * DIM + c]);
  }

  for (int wi = 0; wi < 8; ++wi) {
    const int b = blockIdx.y * 8 + wi;
    const float* xb = x + (size_t)b * DIM * WIN_N;
    if (wi + 1 < 8) __builtin_prefetch(xb + (size_t)DIM * WIN_N, 0, 0);

    v8f acc[4];
    #pragma unroll
    for (int t = 0; t < 4; ++t)
      #pragma unroll
      for (int e = 0; e < 8; ++e) acc[t][e] = 0.f;

    for (int kc = 0; kc < 6; ++kc) {        // K = 384 in chunks of 64
      __syncthreads();                      // Ws ready / Xs reusable
      for (int i = tid; i < 64 * NPAD; i += 128) {
        int k = i >> 6, n = i & 63;         // coalesced global read over n
        float v = (n < WIN_N) ? xb[(size_t)(kc * 64 + k) * WIN_N + n] : 0.f;
        Xs[n * XKP + k] = f2bf(v);          // transpose into token-major
      }
      __syncthreads();

      #pragma unroll
      for (int ks = 0; ks < 2; ++ks) {      // two K=32 WMMA steps
        const int k0 = ks * 32;
        const int ar = wave * 16 + l15;     // A row
        const ushort_t* ap = &Ws[ar * WSP + kc * 64 + k0 + half * 8];
        v16bf A = cat16(ld8(ap), ld8(ap + 16));
        #pragma unroll
        for (int t = 0; t < 4; ++t) {
          const int bn = t * 16 + l15;      // B col (token)
          const ushort_t* bp = &Xs[bn * XKP + k0 + half * 8];
          v16bf Bv = cat16(ld8(bp), ld8(bp + 16));
          acc[t] = wmma_bf16(A, Bv, acc[t]);
        }
      }
    }

    // D layout: VGPR i -> row i + 8*half, col = l15 (+16t)
    #pragma unroll
    for (int t = 0; t < 4; ++t) {
      const int n = t * 16 + l15;
      if (n < WIN_N) {
        #pragma unroll
        for (int i = 0; i < 8; ++i) {
          const int row = M0 + wave * 16 + i + half * 8;
          float v = acc[t][i] + qkv_b[row];
          qkv_o[((size_t)b * QKV_CH + row) * WIN_N + n] = f2bf(v);
        }
      }
    }
  }
}

// ---------------------------------------------------------------------------
// Kernel 2: attention per (window b, head h).  block = 128 (4 waves).
//   S = Q K^T -> scale + rel-pos bias + mask -> softmax -> O = P V
// Qs/Ks token-major [n][DPP]; Vs d-major [d][VSP]; all operand reads are
// two aligned 16B LDS chunks per lane.
// ---------------------------------------------------------------------------
__global__ __launch_bounds__(128) void attn_k(
    const ushort_t* __restrict__ qkv, const float* __restrict__ mask,
    const float* __restrict__ bias_table, ushort_t* __restrict__ o_ws) {
  __shared__ ushort_t Qs[NPAD * DPP];    // [n][d]   5 KB
  __shared__ ushort_t Ks[NPAD * DPP];    // [m][d]   5 KB
  __shared__ ushort_t Vs[HDIM * VSP];    // [d][m]   4.5 KB
  __shared__ float    Sbuf[NPAD * SSP];  // 16.25 KB
  __shared__ ushort_t Pbuf[NPAD * PSP];  // [n][m]   9 KB

  const int b = blockIdx.x, h = blockIdx.y;
  const int tid = threadIdx.x, lane = tid & 31, wave = tid >> 5;
  const int half = lane >> 4, l15 = lane & 15;

  const size_t baseq = ((size_t)b * QKV_CH + h * HDIM) * WIN_N;
  const size_t basek = baseq + (size_t)DIM * WIN_N;
  const size_t basev = baseq + (size_t)(2 * DIM) * WIN_N;
  for (int i = tid; i < HDIM * NPAD; i += 128) {
    int d = i >> 6, n = i & 63;           // coalesced global read over n
    bool ok = (n < WIN_N);
    size_t off = (size_t)d * WIN_N + n;
    ushort_t q = ok ? qkv[baseq + off] : (ushort_t)0;
    ushort_t k = ok ? qkv[basek + off] : (ushort_t)0;
    ushort_t v = ok ? qkv[basev + off] : (ushort_t)0;
    Qs[n * DPP + d] = q;                  // transpose to token-major
    Ks[n * DPP + d] = k;
    Vs[d * VSP + n] = v;                  // keep d-major (contig over n)
  }
  __syncthreads();

  // ---- S = Q K^T : wave owns token rows [16*wave, 16*wave+16) -------------
  {
    v8f s[4];
    #pragma unroll
    for (int t = 0; t < 4; ++t)
      #pragma unroll
      for (int e = 0; e < 8; ++e) s[t][e] = 0.f;

    const int ar = wave * 16 + l15;                 // token row
    const ushort_t* ap = &Qs[ar * DPP + half * 8];  // A[n][d], contig d
    v16bf A = cat16(ld8(ap), ld8(ap + 16));
    #pragma unroll
    for (int t = 0; t < 4; ++t) {
      const int m = t * 16 + l15;                   // B col (token m)
      const ushort_t* bp = &Ks[m * DPP + half * 8];
      v16bf Bv = cat16(ld8(bp), ld8(bp + 16));      // B[d][m] via K[m][d]
      s[t] = wmma_bf16(A, Bv, s[t]);
    }
    #pragma unroll
    for (int t = 0; t < 4; ++t)
      #pragma unroll
      for (int i = 0; i < 8; ++i)
        Sbuf[(wave * 16 + i + half * 8) * SSP + t * 16 + l15] = s[t][i];
  }
  __syncthreads();

  // ---- softmax over m, one row per thread (threads 0..63) -----------------
  if (tid < NPAD) {
    const int n = tid;
    if (n < WIN_N) {
      const int y1 = n / 7, x1 = n % 7;
      const int w = b & (NMASK - 1);
      const float* mrow = mask + ((size_t)w * WIN_N + n) * WIN_N;
      const float sc = 0.17677669529663687f;   // 1/sqrt(32)
      float mx = -1e30f;
      for (int m = 0; m < WIN_N; ++m) {
        int y2 = m / 7, x2 = m % 7;
        int idx = (y1 - y2 + 6) * 13 + (x1 - x2 + 6);
        float v = Sbuf[n * SSP + m] * sc + bias_table[idx * HEADS + h] + mrow[m];
        Sbuf[n * SSP + m] = v;
        mx = fmaxf(mx, v);
      }
      float sum = 0.f;
      for (int m = 0; m < WIN_N; ++m) {
        float e = __expf(Sbuf[n * SSP + m] - mx);
        Sbuf[n * SSP + m] = e;
        sum += e;
      }
      float inv = 1.f / sum;
      for (int m = 0; m < WIN_N; ++m)
        Pbuf[n * PSP + m] = f2bf(Sbuf[n * SSP + m] * inv);
      for (int m = WIN_N; m < NPAD; ++m) Pbuf[n * PSP + m] = 0;
    } else {
      for (int m = 0; m < NPAD; ++m) Pbuf[n * PSP + m] = 0;
    }
  }
  __syncthreads();

  // ---- O = P V : 16 token rows x 32 d-cols per wave, K=64 (2 WMMAs) -------
  v8f o[2];
  #pragma unroll
  for (int t = 0; t < 2; ++t)
    #pragma unroll
    for (int e = 0; e < 8; ++e) o[t][e] = 0.f;

  #pragma unroll
  for (int step = 0; step < 2; ++step) {     // token K-chunks of 32
    const int ar = wave * 16 + l15;
    const int kg = step * 32;
    const ushort_t* ap = &Pbuf[ar * PSP + kg + half * 8];  // A[n][m], contig m
    v16bf A = cat16(ld8(ap), ld8(ap + 16));
    #pragma unroll
    for (int t = 0; t < 2; ++t) {
      const int d = t * 16 + l15;
      const ushort_t* bp = &Vs[d * VSP + kg + half * 8];   // B[m][d] via V[d][m]
      v16bf Bv = cat16(ld8(bp), ld8(bp + 16));
      o[t] = wmma_bf16(A, Bv, o[t]);
    }
  }

  // store channel-major bf16: o_ws[b][h*32+d][n]
  #pragma unroll
  for (int t = 0; t < 2; ++t) {
    const int d = t * 16 + l15;
    #pragma unroll
    for (int i = 0; i < 8; ++i) {
      const int n = wave * 16 + i + half * 8;
      if (n < WIN_N)
        o_ws[((size_t)b * DIM + h * HDIM + d) * WIN_N + n] = f2bf(o[t][i]);
    }
  }
}

// ---------------------------------------------------------------------------
// Kernel 3: output projection.  out = proj_w (384x384) x O[b] (384x49), f32 out
// grid = (6 row-tiles, 512 window-groups of 8), block = 128 (4 waves)
// ---------------------------------------------------------------------------
__global__ __launch_bounds__(128) void proj_gemm_k(
    const ushort_t* __restrict__ o_ws, const float* __restrict__ proj_w,
    float* __restrict__ out) {
  __shared__ ushort_t Ws[64 * WSP];
  __shared__ ushort_t Xs[64 * XKP];   // token-major

  const int tid = threadIdx.x, lane = tid & 31, wave = tid >> 5;
  const int half = lane >> 4, l15 = lane & 15;
  const int M0 = blockIdx.x * 64;

  for (int i = tid; i < 64 * DIM; i += 128) {
    int r = i / DIM, c = i % DIM;
    Ws[r * WSP + c] = f2bf(proj_w[(size_t)(M0 + r) * DIM + c]);
  }

  for (int wi = 0; wi < 8; ++wi) {
    const int b = blockIdx.y * 8 + wi;
    const ushort_t* ob = o_ws + (size_t)b * DIM * WIN_N;
    if (wi + 1 < 8) __builtin_prefetch(ob + (size_t)DIM * WIN_N, 0, 0);

    v8f acc[4];
    #pragma unroll
    for (int t = 0; t < 4; ++t)
      #pragma unroll
      for (int e = 0; e < 8; ++e) acc[t][e] = 0.f;

    for (int kc = 0; kc < 6; ++kc) {
      __syncthreads();
      for (int i = tid; i < 64 * NPAD; i += 128) {
        int k = i >> 6, n = i & 63;
        Xs[n * XKP + k] =
            (n < WIN_N) ? ob[(size_t)(kc * 64 + k) * WIN_N + n] : (ushort_t)0;
      }
      __syncthreads();

      #pragma unroll
      for (int ks = 0; ks < 2; ++ks) {
        const int k0 = ks * 32;
        const int ar = wave * 16 + l15;
        const ushort_t* ap = &Ws[ar * WSP + kc * 64 + k0 + half * 8];
        v16bf A = cat16(ld8(ap), ld8(ap + 16));
        #pragma unroll
        for (int t = 0; t < 4; ++t) {
          const int bn = t * 16 + l15;
          const ushort_t* bp = &Xs[bn * XKP + k0 + half * 8];
          v16bf Bv = cat16(ld8(bp), ld8(bp + 16));
          acc[t] = wmma_bf16(A, Bv, acc[t]);
        }
      }
    }

    #pragma unroll
    for (int t = 0; t < 4; ++t) {
      const int n = t * 16 + l15;
      if (n < WIN_N) {
        #pragma unroll
        for (int i = 0; i < 8; ++i) {
          const int row = M0 + wave * 16 + i + half * 8;
          out[((size_t)b * DIM + row) * WIN_N + n] = acc[t][i];
        }
      }
    }
  }
}

// ---------------------------------------------------------------------------
extern "C" void kernel_launch(void* const* d_in, const int* in_sizes, int n_in,
                              void* d_out, int out_size, void* d_ws, size_t ws_size,
                              hipStream_t stream) {
  (void)in_sizes; (void)n_in; (void)out_size; (void)ws_size;
  const float* x          = (const float*)d_in[0];
  const float* mask       = (const float*)d_in[1];
  const float* qkv_w      = (const float*)d_in[2];
  const float* qkv_b      = (const float*)d_in[3];
  const float* proj_w     = (const float*)d_in[4];
  const float* bias_table = (const float*)d_in[5];
  float* out = (float*)d_out;

  // workspace: bf16 QKV (462 MB) + bf16 attention output (154 MB)
  ushort_t* qkv_ws = (ushort_t*)d_ws;
  const size_t qkv_elems = (size_t)NWIN * QKV_CH * WIN_N;
  ushort_t* o_ws = qkv_ws + qkv_elems;

  qkv_gemm_k<<<dim3(QKV_CH / 64, NWIN / 8), 128, 0, stream>>>(x, qkv_w, qkv_b, qkv_ws);
  attn_k<<<dim3(NWIN, HEADS), 128, 0, stream>>>(qkv_ws, mask, bias_table, o_ws);
  proj_gemm_k<<<dim3(DIM / 64, NWIN / 8), 128, 0, stream>>>(o_ws, proj_w, out);
}
